// SelfAttention_25194278158669
// MI455X (gfx1250) — matrix operand order repaired and checked by reading, not compiled, not run on hardware
//
#include <hip/hip_runtime.h>
#include <hip/hip_bf16.h>

// ---------- types ----------
typedef __attribute__((ext_vector_type(16))) __bf16 v16bf;
typedef __attribute__((ext_vector_type(8)))  __bf16 v8bf;
typedef __attribute__((ext_vector_type(8)))  float  v8f;
typedef __attribute__((ext_vector_type(4)))  float  v4f;

union V16U { v16bf v; v8bf h[2]; __bf16 e[16]; };

// Hardware conversion: clang lowers float->__bf16 fptrunc to the native
// gfx1250 bf16 cvt VALU ops (v_cvt_pk_bf16_f32 class), ~8x cheaper than the
// bit-twiddle RNE sequence.
static __device__ __forceinline__ __bf16 f2bf(float f) {
    return (__bf16)f;
}

static __device__ __forceinline__ v8f wmma_bf16(v16bf a, v16bf b, v8f c) {
    // D = A(16x32 bf16) * B(32x16 bf16) + C(16x16 f32)
    return __builtin_amdgcn_wmma_f32_16x16x32_bf16(
        /*neg_a=*/false, a, /*neg_b=*/false, b,
        /*c_mod=*/(short)0, c, /*reuse_a=*/false, /*reuse_b=*/false);
}

// reduce over the 16-lane half-group (xor masks 1,2,4,8 never cross bit 4)
static __device__ __forceinline__ float rmax16(float v) {
#pragma unroll
    for (int m = 1; m < 16; m <<= 1) v = fmaxf(v, __shfl_xor(v, m, 32));
    return v;
}
static __device__ __forceinline__ float radd16(float v) {
#pragma unroll
    for (int m = 1; m < 16; m <<= 1) v += __shfl_xor(v, m, 32);
    return v;
}

// ---------- constants ----------
#define BB 4
#define SS 4096
#define EE 1024
#define HH 64
#define NEG_BIG (-3.0e38f)

// =====================================================================
// Kernel 0: convert Wq/Wk/Wv (f32 [E,H] row-major) to bf16 in a
// "16-K-chunk" layout: wb[mat][(k>>4)*H*16 + n*16 + (k&15)].
// A lane's B-matrix fragment (16 consecutive K for one column n) is then
// one contiguous 32-byte load.
// =====================================================================
__global__ void __launch_bounds__(256)
cvt_w_kernel(const float* __restrict__ Wq, const float* __restrict__ Wk,
             const float* __restrict__ Wv, __bf16* __restrict__ wb) {
    int id = blockIdx.x * 256 + threadIdx.x;       // 0 .. 3*E*H-1 (exact grid)
    int mat = id >> 16;                            // E*H = 65536 per matrix
    int rem = id & 65535;
    int k = rem >> 6;                              // 0..1023
    int n = rem & 63;                              // 0..63
    const float* W = (mat == 0) ? Wq : (mat == 1) ? Wk : Wv;
    float v = W[k * HH + n];
    wb[mat * 65536 + (k >> 4) * (HH * 16) + n * 16 + (k & 15)] = f2bf(v);
}

// =====================================================================
// Kernel 1: QKV projection. One wave per 16-row tile of x.
// q -> bf16 [B,S,H] pre-scaled by H^-0.5, k -> bf16 [B,S,H],
// v -> bf16 transposed [B,H,S] (so P*V B-fragments are contiguous).
// =====================================================================
__global__ void __launch_bounds__(128)
qkv_kernel(const float* __restrict__ x, const __bf16* __restrict__ wb,
           __bf16* __restrict__ qb, __bf16* __restrict__ kb,
           __bf16* __restrict__ vT) {
    int lane = threadIdx.x & 31;
    int warp = threadIdx.x >> 5;
    int tile = blockIdx.x * 4 + warp;              // 0..1023
    int b  = tile >> 8;
    int s0 = (tile & 255) << 4;
    int m  = lane & 15;                            // row within tile / column n
    int kh = lane >> 4;                            // K-half selector

    v8f acc[3][4];
#pragma unroll
    for (int mt = 0; mt < 3; ++mt)
#pragma unroll
        for (int t = 0; t < 4; ++t) acc[mt][t] = (v8f)0.0f;

    const float* xrow = x + ((size_t)(b * SS + s0 + m)) * EE;

    for (int ks = 0; ks < EE / 32; ++ks) {
        int k0 = ks * 32;
        // A fragment: row m, elems 0..15 = K kh*16+0..15
        const float* xp = xrow + k0 + kh * 16;
        v4f a0 = *(const v4f*)(xp);
        v4f a1 = *(const v4f*)(xp + 4);
        v4f a2 = *(const v4f*)(xp + 8);
        v4f a3 = *(const v4f*)(xp + 12);
        V16U A;
#pragma unroll
        for (int j = 0; j < 4; ++j) {
            A.e[j]      = f2bf(a0[j]);
            A.e[4 + j]  = f2bf(a1[j]);
            A.e[8 + j]  = f2bf(a2[j]);
            A.e[12 + j] = f2bf(a3[j]);
        }
        // B fragments: contiguous 32B per lane from chunked layout
#pragma unroll
        for (int mt = 0; mt < 3; ++mt) {
#pragma unroll
            for (int t = 0; t < 4; ++t) {
                const v16bf Bv = *(const v16bf*)(wb + mt * 65536 +
                                                 (ks * 2 + kh) * (HH * 16) +
                                                 (t * 16 + m) * 16);
                acc[mt][t] = wmma_bf16(A.v, Bv, acc[mt][t]);
            }
        }
    }

    // Store q/k from C layout: VGPR r -> row s0 + r + 8*kh, column t*16 + m
#pragma unroll
    for (int t = 0; t < 4; ++t) {
#pragma unroll
        for (int r = 0; r < 8; ++r) {
            int s = s0 + r + 8 * kh;
            int h = t * 16 + m;
            qb[(size_t)(b * SS + s) * HH + h] = f2bf(acc[0][t][r] * 0.125f);
            kb[(size_t)(b * SS + s) * HH + h] = f2bf(acc[1][t][r]);
        }
    }
    // Store v transposed, packed: a lane's 8 accumulator entries for one
    // n-tile are 8 *consecutive s* for a fixed h -> one b128 store each.
#pragma unroll
    for (int t = 0; t < 4; ++t) {
        v8bf pv;
#pragma unroll
        for (int r = 0; r < 8; ++r) pv[r] = f2bf(acc[2][t][r]);
        int h = t * 16 + m;
        *(v8bf*)(vT + (size_t)(b * HH + h) * SS + s0 + 8 * kh) = pv;
    }
}

// =====================================================================
// Kernel 2: causal flash attention. One wave per 16-row q-tile,
// key chunks of 32. Q*K^T: 4 WMMAs/chunk; P*V: 4 WMMAs/chunk.
// Online softmax with 16-lane shuffle reductions; P re-layout via LDS.
// =====================================================================
__global__ void __launch_bounds__(128)
attn_kernel(const __bf16* __restrict__ qb, const __bf16* __restrict__ kb,
            const __bf16* __restrict__ vT, float* __restrict__ out) {
    __shared__ __bf16 lds[4][16 * 32];             // per-wave P staging tile

    int lane = threadIdx.x & 31;
    int warp = threadIdx.x >> 5;
    int tile = blockIdx.x * 4 + warp;              // 0..1023
    int b  = tile >> 8;
    int q0 = (tile & 255) << 4;
    int m  = lane & 15;
    int kh = lane >> 4;
    __bf16* P = lds[warp];

    // Q A-fragments for the two 32-wide h-chunks (kept in regs all loop long)
    V16U Aq[2];
#pragma unroll
    for (int c = 0; c < 2; ++c) {
        const __bf16* qp = qb + ((size_t)(b * SS + q0 + m)) * HH + c * 32 + kh * 16;
        Aq[c].h[0] = *(const v8bf*)(qp);
        Aq[c].h[1] = *(const v8bf*)(qp + 8);
    }

    v8f O[4];
#pragma unroll
    for (int t = 0; t < 4; ++t) O[t] = (v8f)0.0f;
    float mr[8], lr[8];
#pragma unroll
    for (int r = 0; r < 8; ++r) { mr[r] = NEG_BIG; lr[r] = 0.0f; }

    const int kend = q0 + 16;
    for (int base = 0; base < kend; base += 32) {
        // ---- Q * K^T : two 16x16 att tiles (keys base..+15, base+16..+31)
        v8f c0 = (v8f)0.0f, c1 = (v8f)0.0f;
#pragma unroll
        for (int kk = 0; kk < 2; ++kk) {           // reduction over h halves
            const v16bf B0 = *(const v16bf*)(kb +
                ((size_t)(b * SS + base + m)) * HH + kk * 32 + kh * 16);
            const v16bf B1 = *(const v16bf*)(kb +
                ((size_t)(b * SS + base + 16 + m)) * HH + kk * 32 + kh * 16);
            c0 = wmma_bf16(Aq[kk].v, B0, c0);
            c1 = wmma_bf16(Aq[kk].v, B1, c1);
        }

        const bool needmask = (base + 31 > q0);    // chunk touches the diagonal

        // ---- online softmax per q-row (row r+8*kh lives in 16-lane half-group)
#pragma unroll
        for (int r = 0; r < 8; ++r) {
            int row = q0 + r + 8 * kh;
            float a0 = c0[r], a1 = c1[r];
            if (needmask) {
                a0 = (base + m      > row) ? NEG_BIG : a0;
                a1 = (base + 16 + m > row) ? NEG_BIG : a1;
            }
            float mx = rmax16(fmaxf(a0, a1));
            float mn = fmaxf(mr[r], mx);
            float sc = __expf(mr[r] - mn);
            float p0 = __expf(a0 - mn);
            float p1 = __expf(a1 - mn);
            lr[r] = lr[r] * sc + radd16(p0 + p1);
            mr[r] = mn;
#pragma unroll
            for (int t = 0; t < 4; ++t) O[t][r] *= sc;
            // stage P (C layout -> LDS row-major [16 rows][32 keys])
            P[(r + 8 * kh) * 32 + m]      = f2bf(p0);
            P[(r + 8 * kh) * 32 + 16 + m] = f2bf(p1);
        }
        // DS ops of one wave execute in order; explicit wait is belt & braces
        asm volatile("s_wait_dscnt 0" ::: "memory");

        // ---- reload P in A-matrix layout
        V16U Ap;
        Ap.h[0] = *(const v8bf*)&P[m * 32 + kh * 16];
        Ap.h[1] = *(const v8bf*)&P[m * 32 + kh * 16 + 8];

        // ---- P * V : 4 n-tiles over h, B from transposed V (contiguous)
#pragma unroll
        for (int t = 0; t < 4; ++t) {
            const v16bf Bv = *(const v16bf*)(vT +
                ((size_t)(b * HH + t * 16 + m)) * SS + base + kh * 16);
            O[t] = wmma_bf16(Ap.v, Bv, O[t]);
        }
    }

    // ---- normalize and store f32 output [B,S,H]
#pragma unroll
    for (int t = 0; t < 4; ++t) {
#pragma unroll
        for (int r = 0; r < 8; ++r) {
            int s = q0 + r + 8 * kh;
            out[((size_t)(b * SS + s)) * HH + t * 16 + m] = O[t][r] / lr[r];
        }
    }
}

// =====================================================================
// Host-side launcher
// =====================================================================
extern "C" void kernel_launch(void* const* d_in, const int* in_sizes, int n_in,
                              void* d_out, int out_size, void* d_ws, size_t ws_size,
                              hipStream_t stream) {
    const float* x  = (const float*)d_in[0];   // [B,S,E]
    const float* Wq = (const float*)d_in[1];   // [E,H]
    const float* Wk = (const float*)d_in[2];
    const float* Wv = (const float*)d_in[3];

    // workspace layout (bf16): qb | kb | vT | wb   (~6.4 MiB total)
    __bf16* qb = (__bf16*)d_ws;
    __bf16* kb = qb + (size_t)BB * SS * HH;        // +1,048,576
    __bf16* vT = kb + (size_t)BB * SS * HH;
    __bf16* wb = vT + (size_t)BB * SS * HH;        // 3*E*H = 196,608

    cvt_w_kernel<<<(3 * EE * HH) / 256, 256, 0, stream>>>(Wq, Wk, Wv, wb);
    qkv_kernel<<<(BB * SS / 16) / 4, 128, 0, stream>>>(x, wb, qb, kb, vT);
    attn_kernel<<<(BB * SS / 16) / 4, 128, 0, stream>>>(qb, kb, vT, (float*)d_out);
}
